// HandWritingSynthesisNet_35665408426178
// MI455X (gfx1250) — matrix operand order repaired and checked by reading, not compiled, not run on hardware
//
#include <hip/hip_runtime.h>
#include <hip/hip_bf16.h>
#include <stdint.h>

// ---------------------------------------------------------------------------
// Handwriting-synthesis net on MI455X (gfx1250, wave32, WMMA bf16).
//
// Plan (roofline-driven):
//  * Recurrence (3 layers x 600 steps) is latency bound (85 MFLOP/step) ->
//    persistent kernels: 25 WGs x 256 thr, W_hh slice resident in LDS,
//    h exchanged via small L2-hot double buffer + software grid barrier.
//  * Input contributions for layers 2/3 and output projection are big
//    parallel GEMMs (~130 GFLOP, compute bound) -> tiled bf16 WMMA
//    (16x16x32, f32 acc) with 2x4 tiles per wave (0.75 frag-loads/WMMA).
//  * All GEMM operands kept in bf16 (ushort) to halve HBM traffic; the
//    precomputed gate buffer `pre` stays f32 (246 MB ~ 21us @ 23.3 TB/s).
// ---------------------------------------------------------------------------

#define B_   64
#define T_   600
#define H_   400
#define V_   77
#define O_   121
#define U_   64
#define G_   1600          // 4*H
#define BT   (B_*T_)       // 38400
#define KP1  480           // layer1 K: [w(77) | h(400) | pad(3)]
#define KP23 416           // layer2/3 K: h(400) padded to 13*32
#define KPO  1248          // output K: 3 * 416
#define NWG  25            // persistent workgroups (16 hidden units each)

typedef __attribute__((ext_vector_type(16))) __bf16         v16bf;
typedef __attribute__((ext_vector_type(16))) unsigned short v16u;
typedef __attribute__((ext_vector_type(8)))  float          v8f;

__device__ inline unsigned short f2bfu(float f) {
  unsigned u = __float_as_uint(f);
  return (unsigned short)((u + 0x7FFFu + ((u >> 16) & 1u)) >> 16); // RNE
}
__device__ inline float bf2f(unsigned short h) {
  return __uint_as_float(((unsigned)h) << 16);
}
__device__ inline float fsig(float x)  { return 1.f / (1.f + __expf(-x)); }
__device__ inline float ftanh(float x) {
  x = fminf(fmaxf(x, -15.f), 15.f);
  float e = __expf(2.f * x);
  return (e - 1.f) / (e + 1.f);
}

// Two 8-ushort (16B) chunks -> one 16-element bf16 fragment.
__device__ inline v16u ld_frag16(const unsigned short* p0, const unsigned short* p1) {
  union { v16u v; uint4 q[2]; } u;
  u.q[0] = *(const uint4*)p0;
  u.q[1] = *(const uint4*)p1;
  return u.v;
}
// A fragment: 16x32 (MxK) row-major tile.  Per ISA layout: lane m=l&15,
// hi=l>>4; halfs 0..7 -> K = hi*8+j ; halfs 8..15 -> K = 16+hi*8+j.
__device__ inline v16u ld_afrag(const unsigned short* A, int lda, int lane) {
  int m = lane & 15, hi = lane >> 4;
  const unsigned short* r = A + m * lda + hi * 8;
  return ld_frag16(r, r + 16);
}
// B^T fragment from row-major W (N x K): lane n=l&15 reads row n,
// halfs 0..7 -> K = hi*8+j ; halfs 8..15 -> K = 16+hi*8+j (same as A).
__device__ inline v16u ld_bfragT(const unsigned short* Wrow0, int ldw, int lane) {
  int n = lane & 15, hi = lane >> 4;
  const unsigned short* r = Wrow0 + n * ldw + hi * 8;
  return ld_frag16(r, r + 16);
}
__device__ inline v8f wmma_bf16(v16u a, v16u b, v8f c) {
  union { v16u u; v16bf h; } A, Bv;
  A.u = a; Bv.u = b;
  return __builtin_amdgcn_wmma_f32_16x16x32_bf16(
      false, A.h, false, Bv.h, (short)0, c, false, false);
}

// Software grid barrier (sense via monotonically increasing generation).
// Requires all NWG blocks co-resident (25 blocks -> trivially true).
__device__ inline void grid_sync(int* bar, int nwg) {
  __threadfence();
  __syncthreads();
  if (threadIdx.x == 0) {
    volatile int* vgen = bar + 1;
    int g = *vgen;
    if (atomicAdd(bar, 1) == nwg - 1) {
      *(volatile int*)bar = 0;
      __threadfence();
      atomicAdd(bar + 1, 1);
    } else {
      while (*vgen == g) __builtin_amdgcn_s_sleep(2);
    }
  }
  __syncthreads();
  __threadfence();
}

// ---------------------------------------------------------------------------
// Weight conversion / packing kernels (f32 -> padded bf16)
// ---------------------------------------------------------------------------
__global__ __launch_bounds__(256) void cvt_pad_kernel(
    const float* __restrict__ W, unsigned short* __restrict__ Wb,
    int N, int K, int KP) {
  long i = (long)blockIdx.x * 256 + threadIdx.x;
  if (i >= (long)N * KP) return;
  int n = (int)(i / KP), k = (int)(i % KP);
  Wb[i] = (k < K) ? f2bfu(W[(long)n * K + k]) : (unsigned short)0;
}

// Wcat1 (1600 x 480): cols 0..76 = W_ih1[:,3..79], 77..476 = W_hh1, rest 0.
__global__ __launch_bounds__(256) void cvt_wcat1_kernel(
    const float* __restrict__ Wih1, const float* __restrict__ Whh1,
    unsigned short* __restrict__ Wb) {
  long i = (long)blockIdx.x * 256 + threadIdx.x;
  if (i >= (long)G_ * KP1) return;
  int n = (int)(i / KP1), c = (int)(i % KP1);
  float v = 0.f;
  if (c < V_)            v = Wih1[n * 80 + 3 + c];
  else if (c < V_ + H_)  v = Whh1[n * H_ + (c - V_)];
  Wb[i] = f2bfu(v);
}

// Wout (128 x 1248): row n<121, segment s: cols s*416+k (k<400) = W_out[n, s*400+k]
__global__ __launch_bounds__(256) void cvt_wout_kernel(
    const float* __restrict__ Wo, unsigned short* __restrict__ Wb) {
  long i = (long)blockIdx.x * 256 + threadIdx.x;
  if (i >= 128L * KPO) return;
  int n = (int)(i / KPO), c = (int)(i % KPO);
  int s = c / KP23, k = c % KP23;
  float v = 0.f;
  if (n < O_ && k < H_) v = Wo[(long)n * 1200 + s * H_ + k];
  Wb[i] = f2bfu(v);
}

// Xcat (BT x 480) = [x(3) | h_prev(400) | w(77)] in bf16.
__global__ __launch_bounds__(256) void build_xcat_kernel(
    const float* __restrict__ x, const unsigned short* __restrict__ hprev,
    const unsigned short* __restrict__ wb, unsigned short* __restrict__ Xc) {
  long i = (long)blockIdx.x * 256 + threadIdx.x;
  if (i >= (long)BT * KP1) return;
  long r = i / KP1; int c = (int)(i % KP1);
  unsigned short v;
  if (c < 3)        v = f2bfu(x[r * 3 + c]);
  else if (c < 403) v = hprev[r * KP23 + (c - 3)];
  else              v = wb[r * V_ + (c - 403)];
  Xc[i] = v;
}

// ---------------------------------------------------------------------------
// Big parallel GEMM: C(MxN,f32) = A(MxKP,bf16) * W(NxKP,bf16)^T
// WG tile 256(M) x 64(N): 8 waves, each wave = 2 m-tiles x 4 n-tiles
// (8 f32 accumulators, 6 fragment loads per 8 WMMAs).
// ---------------------------------------------------------------------------
__global__ __launch_bounds__(256) void gemm_nt_kernel(
    const unsigned short* __restrict__ A, const unsigned short* __restrict__ W,
    float* __restrict__ C, int N, int KP, int NKT) {
  int mb = blockIdx.x * 256, nb = blockIdx.y * 64;
  int wv = threadIdx.x >> 5, lane = threadIdx.x & 31;
  const unsigned short* A0 = A + (long)(mb + wv * 32) * KP;
  const unsigned short* A1 = A0 + (long)16 * KP;
  const unsigned short* W0 = W + (long)nb * KP;
  v8f z = {0,0,0,0,0,0,0,0};
  v8f acc[2][4];
#pragma unroll
  for (int i = 0; i < 2; ++i)
#pragma unroll
    for (int j = 0; j < 4; ++j) acc[i][j] = z;
  for (int kt = 0; kt < NKT; ++kt) {
    v16u a0 = ld_afrag(A0 + kt * 32, KP, lane);
    v16u a1 = ld_afrag(A1 + kt * 32, KP, lane);
#pragma unroll
    for (int j = 0; j < 4; ++j) {
      v16u b = ld_bfragT(W0 + (long)j * 16 * KP + kt * 32, KP, lane);
      acc[0][j] = wmma_bf16(a0, b, acc[0][j]);
      acc[1][j] = wmma_bf16(a1, b, acc[1][j]);
    }
  }
  int hi = lane >> 4, n = lane & 15;
#pragma unroll
  for (int i = 0; i < 2; ++i)
#pragma unroll
    for (int j = 0; j < 4; ++j)
#pragma unroll
      for (int r = 0; r < 8; ++r) {
        long m = mb + wv * 32 + i * 16 + hi * 8 + r;
        C[m * N + nb + j * 16 + n] = acc[i][j][r];
      }
}

// Output GEMM: out(BT x 121) = [h1|h2|h3](bf16, padded 416 each) * Wo^T + bo
// Same 256x64 WG tile / 2x4 wave tile; K runs over 3 padded segments.
__global__ __launch_bounds__(256) void gemm_out_kernel(
    const unsigned short* __restrict__ h1, const unsigned short* __restrict__ h2,
    const unsigned short* __restrict__ h3, const unsigned short* __restrict__ Wo,
    const float* __restrict__ bo, float* __restrict__ out) {
  int mb = blockIdx.x * 256, nb = blockIdx.y * 64;
  int wv = threadIdx.x >> 5, lane = threadIdx.x & 31;
  v8f z = {0,0,0,0,0,0,0,0};
  v8f acc[2][4];
#pragma unroll
  for (int i = 0; i < 2; ++i)
#pragma unroll
    for (int j = 0; j < 4; ++j) acc[i][j] = z;
  const unsigned short* seg[3] = {h1, h2, h3};
  for (int s = 0; s < 3; ++s) {
    const unsigned short* A0 = seg[s] + (long)(mb + wv * 32) * KP23;
    const unsigned short* A1 = A0 + (long)16 * KP23;
    const unsigned short* W0 = Wo + (long)nb * KPO + s * KP23;
    for (int kt = 0; kt < 13; ++kt) {
      v16u a0 = ld_afrag(A0 + kt * 32, KP23, lane);
      v16u a1 = ld_afrag(A1 + kt * 32, KP23, lane);
#pragma unroll
      for (int j = 0; j < 4; ++j) {
        v16u b = ld_bfragT(W0 + (long)j * 16 * KPO + kt * 32, KPO, lane);
        acc[0][j] = wmma_bf16(a0, b, acc[0][j]);
        acc[1][j] = wmma_bf16(a1, b, acc[1][j]);
      }
    }
  }
  int hi = lane >> 4, n = lane & 15;
#pragma unroll
  for (int i = 0; i < 2; ++i)
#pragma unroll
    for (int j = 0; j < 4; ++j)
#pragma unroll
      for (int r = 0; r < 8; ++r) {
        long m = mb + wv * 32 + i * 16 + hi * 8 + r;
        int c = nb + j * 16 + n;
        if (c < O_) out[m * O_ + c] = acc[i][j][r] + bo[c];
      }
}

// ---------------------------------------------------------------------------
// Persistent LSTM layers 2/3: per step g = h_{t-1} @ Whh^T (WMMA, LDS weights)
//            + pre[t] + bias; pointwise -> c,h; grid barrier between steps.
// Each of 25 WGs owns 16 hidden units (=> 64 gate columns = 4 n-tiles).
// ---------------------------------------------------------------------------
__global__ __launch_bounds__(256) void lstm_rnn_kernel(
    const float* __restrict__ pre,        // (BT x 1600)
    const unsigned short* __restrict__ Whh,// (1600 x 416) bf16 padded
    const float* __restrict__ bias,       // (1600)
    unsigned short* __restrict__ hout,    // (BT x 416) bf16, pads pre-zeroed
    unsigned short* hbuf0, unsigned short* hbuf1, // (64 x 416) each, zeroed
    int* bar) {
  extern __shared__ char smem[];
  unsigned short* Wl = (unsigned short*)smem;                   // 64 x 416
  float* Gl = (float*)(smem + 64 * KP23 * 2);                   // 64 x 64
  float* Cs = (float*)(smem + 64 * KP23 * 2 + 64 * 64 * 4);     // 64 x 16
  const int u0 = blockIdx.x * 16;
  const int tid = threadIdx.x;

  // Stage this WG's weight slice into LDS (rows: gate g, unit n -> g*400+u0+n)
  for (int i = tid; i < 64 * (KP23 / 8); i += 256) {
    int row = i / (KP23 / 8), c8 = i % (KP23 / 8);
    int g = row >> 4, n = row & 15;
    ((uint4*)Wl)[row * (KP23 / 8) + c8] =
        ((const uint4*)(Whh + (long)(g * H_ + u0 + n) * KP23))[c8];
  }
  for (int i = tid; i < 64 * 16; i += 256) Cs[i] = 0.f;
  __syncthreads();

  const int wv = tid >> 5, lane = tid & 31;
  const int mt = wv & 3, ng = (wv >> 2) * 2;

  for (int t = 0; t < T_; ++t) {
    const unsigned short* A = (t & 1) ? hbuf1 : hbuf0;
    unsigned short* Anext   = (t & 1) ? hbuf0 : hbuf1;
    v8f acc0 = {0,0,0,0,0,0,0,0}, acc1 = {0,0,0,0,0,0,0,0};
    const unsigned short* Ab = A + mt * 16 * KP23;
    for (int kt = 0; kt < 13; ++kt) {
      v16u a  = ld_afrag (Ab + kt * 32, KP23, lane);
      v16u b0 = ld_bfragT(Wl + (ng * 16) * KP23 + kt * 32, KP23, lane);
      v16u b1 = ld_bfragT(Wl + (ng * 16 + 16) * KP23 + kt * 32, KP23, lane);
      acc0 = wmma_bf16(a, b0, acc0);
      acc1 = wmma_bf16(a, b1, acc1);
    }
    int hi = lane >> 4, n = lane & 15;
#pragma unroll
    for (int r = 0; r < 8; ++r) {
      int m = mt * 16 + hi * 8 + r;
      Gl[m * 64 + ng * 16 + n]       = acc0[r];
      Gl[m * 64 + (ng + 1) * 16 + n] = acc1[r];
    }
    __syncthreads();
    for (int e = tid; e < 64 * 16; e += 256) {
      int b = e >> 4, u = e & 15;
      const float* pr = pre + ((long)(b * T_ + t)) * G_ + u0 + u;
      float gi = Gl[b * 64 + u]        + pr[0]    + bias[u0 + u];
      float gf = Gl[b * 64 + 16 + u]   + pr[400]  + bias[400 + u0 + u];
      float gg = Gl[b * 64 + 32 + u]   + pr[800]  + bias[800 + u0 + u];
      float go = Gl[b * 64 + 48 + u]   + pr[1200] + bias[1200 + u0 + u];
      float c = fsig(gf) * Cs[e] + fsig(gi) * ftanh(gg);
      float h = fsig(go) * ftanh(c);
      Cs[e] = c;
      unsigned short hb = f2bfu(h);
      Anext[b * KP23 + u0 + u] = hb;
      hout[((long)(b * T_ + t)) * KP23 + u0 + u] = hb;
    }
    grid_sync(bar, NWG);
  }
}

// ---------------------------------------------------------------------------
// Persistent LSTM layer 1 + Gaussian attention window.
// A_t = [w_{t-1}(77) | h_{t-1}(400) | pad(3)] (64 x 480 bf16, double-buffered)
// g = A_t @ Wcat1^T (WMMA) + x_t @ W_ih1[:, :3]^T + b1 (inline)
// Phase2 (WGs 0..15, 4 batches each): mix=exp(h@Wwin^T+bwin), kappa+=,
//   phi over U=64, w = phi . one_hot(text)  (deterministic per-char scan).
// ---------------------------------------------------------------------------
__global__ __launch_bounds__(256) void lstm1_kernel(
    const float* __restrict__ xin,          // (B,T,3)
    const int* __restrict__ text,           // (B,U)
    const float* __restrict__ tmask,        // (B,U)
    const unsigned short* __restrict__ Wcat1,// (1600 x 480) bf16
    const float* __restrict__ Wih1,         // (1600 x 80) f32 (x columns)
    const float* __restrict__ b1,
    const float* __restrict__ Wwin,         // (30 x 400)
    const float* __restrict__ bwin,         // (30)
    float* __restrict__ kap,                // (B x 10) state, zeroed
    unsigned short* __restrict__ hout,      // (BT x 416) bf16
    unsigned short* __restrict__ wb,        // (BT x 77) bf16
    unsigned short* abuf0, unsigned short* abuf1, // (64 x 480), zeroed
    int* bar) {
  extern __shared__ char smem[];
  unsigned short* Wl = (unsigned short*)smem;                 // 64 x 480
  float* Gl   = (float*)(smem + 64 * KP1 * 2);                // 64 x 64
  float* Cs   = (float*)(smem + 64 * KP1 * 2 + 16384);        // 64 x 16
  float* mixs = (float*)(smem + 64 * KP1 * 2 + 16384 + 4096); // 4 x 32
  float* kapl = mixs + 4 * 32;                                // 4 x 16
  float* phis = kapl + 4 * 16;                                // 4 x 64
  const int wg = blockIdx.x;
  const int u0 = wg * 16;
  const int tid = threadIdx.x;

  for (int i = tid; i < 64 * (KP1 / 8); i += 256) {
    int row = i / (KP1 / 8), c8 = i % (KP1 / 8);
    int g = row >> 4, n = row & 15;
    ((uint4*)Wl)[row * (KP1 / 8) + c8] =
        ((const uint4*)(Wcat1 + (long)(g * H_ + u0 + n) * KP1))[c8];
  }
  for (int i = tid; i < 64 * 16; i += 256) Cs[i] = 0.f;
  __syncthreads();

  const int wv = tid >> 5, lane = tid & 31;
  const int mt = wv & 3, ng = (wv >> 2) * 2;

  for (int t = 0; t < T_; ++t) {
    const unsigned short* A = (t & 1) ? abuf1 : abuf0;
    unsigned short* Anext   = (t & 1) ? abuf0 : abuf1;
    v8f acc0 = {0,0,0,0,0,0,0,0}, acc1 = {0,0,0,0,0,0,0,0};
    const unsigned short* Ab = A + mt * 16 * KP1;
    for (int kt = 0; kt < 15; ++kt) {
      v16u a  = ld_afrag (Ab + kt * 32, KP1, lane);
      v16u b0 = ld_bfragT(Wl + (ng * 16) * KP1 + kt * 32, KP1, lane);
      v16u b1v= ld_bfragT(Wl + (ng * 16 + 16) * KP1 + kt * 32, KP1, lane);
      acc0 = wmma_bf16(a, b0, acc0);
      acc1 = wmma_bf16(a, b1v, acc1);
    }
    int hi = lane >> 4, n = lane & 15;
#pragma unroll
    for (int r = 0; r < 8; ++r) {
      int m = mt * 16 + hi * 8 + r;
      Gl[m * 64 + ng * 16 + n]       = acc0[r];
      Gl[m * 64 + (ng + 1) * 16 + n] = acc1[r];
    }
    __syncthreads();
    for (int e = tid; e < 64 * 16; e += 256) {
      int b = e >> 4, u = e & 15;
      const float* xr = xin + (long)(b * T_ + t) * 3;
      int r0 = u0 + u;
      float gi = Gl[b * 64 + u]      + b1[r0] +
                 xr[0]*Wih1[r0*80] + xr[1]*Wih1[r0*80+1] + xr[2]*Wih1[r0*80+2];
      int r1 = 400 + r0;
      float gf = Gl[b * 64 + 16 + u] + b1[r1] +
                 xr[0]*Wih1[r1*80] + xr[1]*Wih1[r1*80+1] + xr[2]*Wih1[r1*80+2];
      int r2 = 800 + r0;
      float gg = Gl[b * 64 + 32 + u] + b1[r2] +
                 xr[0]*Wih1[r2*80] + xr[1]*Wih1[r2*80+1] + xr[2]*Wih1[r2*80+2];
      int r3 = 1200 + r0;
      float go = Gl[b * 64 + 48 + u] + b1[r3] +
                 xr[0]*Wih1[r3*80] + xr[1]*Wih1[r3*80+1] + xr[2]*Wih1[r3*80+2];
      float c = fsig(gf) * Cs[e] + fsig(gi) * ftanh(gg);
      float h = fsig(go) * ftanh(c);
      Cs[e] = c;
      unsigned short hb = f2bfu(h);
      Anext[b * KP1 + V_ + u0 + u] = hb;
      hout[((long)(b * T_ + t)) * KP23 + u0 + u] = hb;
    }
    grid_sync(bar, NWG);   // all h slices visible

    // ---- attention phase: WGs 0..15, 4 batches each ----
    if (wg < 16) {
      int grp = tid >> 6, lt = tid & 63;
      int b = wg * 4 + grp;
      if (lt < 30) {
        float s = bwin[lt];
        const unsigned short* hr = hout + ((long)(b * T_ + t)) * KP23;
        for (int k = 0; k < H_; ++k) s += bf2f(hr[k]) * Wwin[lt * H_ + k];
        mixs[grp * 32 + lt] = __expf(s);
      }
      __syncthreads();
      if (lt < 10) {
        float kn = kap[b * 10 + lt] + mixs[grp * 32 + 20 + lt];
        kap[b * 10 + lt] = kn;
        kapl[grp * 16 + lt] = kn;
      }
      __syncthreads();
      {
        float uu = (float)lt, phi = 0.f;
#pragma unroll
        for (int k = 0; k < 10; ++k) {
          float d = kapl[grp * 16 + k] - uu;
          phi += mixs[grp * 32 + k] * __expf(-mixs[grp * 32 + 10 + k] * d * d);
        }
        phis[grp * 64 + lt] = phi * tmask[b * U_ + lt];
      }
      __syncthreads();
      for (int v = lt; v < V_; v += 64) {
        float s = 0.f;
        for (int u = 0; u < U_; ++u)
          if (text[b * U_ + u] == v) s += phis[grp * 64 + u];
        unsigned short w16 = f2bfu(s);
        Anext[b * KP1 + v] = w16;
        wb[((long)(b * T_ + t)) * V_ + v] = w16;
      }
    }
    grid_sync(bar, NWG);   // w_t visible for step t+1 readers
  }
}

// ---------------------------------------------------------------------------
extern "C" void kernel_launch(void* const* d_in, const int* in_sizes, int n_in,
                              void* d_out, int out_size, void* d_ws, size_t ws_size,
                              hipStream_t stream) {
  const float* x    = (const float*)d_in[0];
  const int*   text = (const int*)  d_in[1];
  const float* mask = (const float*)d_in[2];
  const float* Wih1 = (const float*)d_in[3];
  const float* Whh1 = (const float*)d_in[4];
  const float* b1   = (const float*)d_in[5];
  const float* Wih2 = (const float*)d_in[6];
  const float* Whh2 = (const float*)d_in[7];
  const float* b2   = (const float*)d_in[8];
  const float* Wih3 = (const float*)d_in[9];
  const float* Whh3 = (const float*)d_in[10];
  const float* b3   = (const float*)d_in[11];
  const float* Wwin = (const float*)d_in[12];
  const float* bwin = (const float*)d_in[13];
  const float* Wout = (const float*)d_in[14];
  const float* bout = (const float*)d_in[15];
  float* out = (float*)d_out;

  char* p = (char*)d_ws;
  auto alloc = [&](size_t bytes) -> char* {
    char* r = p; p += (bytes + 255) & ~(size_t)255; return r;
  };
  // --- zero-initialized zone (single memset) ---
  int*            bar  = (int*)           alloc(256);
  float*          kap  = (float*)         alloc(B_ * 10 * 4);
  unsigned short* A1a  = (unsigned short*)alloc(64 * KP1 * 2);
  unsigned short* A1b  = (unsigned short*)alloc(64 * KP1 * 2);
  unsigned short* A2a  = (unsigned short*)alloc(64 * KP23 * 2);
  unsigned short* A2b  = (unsigned short*)alloc(64 * KP23 * 2);
  unsigned short* A3a  = (unsigned short*)alloc(64 * KP23 * 2);
  unsigned short* A3b  = (unsigned short*)alloc(64 * KP23 * 2);
  unsigned short* h1b  = (unsigned short*)alloc((size_t)BT * KP23 * 2);
  unsigned short* h2b  = (unsigned short*)alloc((size_t)BT * KP23 * 2);
  unsigned short* h3b  = (unsigned short*)alloc((size_t)BT * KP23 * 2);
  size_t zero_bytes = (size_t)(p - (char*)d_ws);
  // --- fully-overwritten zone ---
  unsigned short* wbuf = (unsigned short*)alloc((size_t)BT * V_ * 2);
  unsigned short* Xcat = (unsigned short*)alloc((size_t)BT * KP1 * 2);
  float*          pre  = (float*)         alloc((size_t)BT * G_ * 4);
  unsigned short* Wc1  = (unsigned short*)alloc((size_t)G_ * KP1 * 2);
  unsigned short* Wh2  = (unsigned short*)alloc((size_t)G_ * KP23 * 2);
  unsigned short* Wh3  = (unsigned short*)alloc((size_t)G_ * KP23 * 2);
  unsigned short* Wi2  = (unsigned short*)alloc((size_t)G_ * KP1 * 2);
  unsigned short* Wi3  = (unsigned short*)alloc((size_t)G_ * KP1 * 2);
  unsigned short* Wob  = (unsigned short*)alloc((size_t)128 * KPO * 2);

  hipMemsetAsync(d_ws, 0, zero_bytes, stream);

  // Weight packing
  cvt_wcat1_kernel<<<(G_*KP1 + 255)/256, 256, 0, stream>>>(Wih1, Whh1, Wc1);
  cvt_pad_kernel<<<(G_*KP23 + 255)/256, 256, 0, stream>>>(Whh2, Wh2, G_, H_, KP23);
  cvt_pad_kernel<<<(G_*KP23 + 255)/256, 256, 0, stream>>>(Whh3, Wh3, G_, H_, KP23);
  cvt_pad_kernel<<<(G_*KP1  + 255)/256, 256, 0, stream>>>(Wih2, Wi2, G_, KP1, KP1);
  cvt_pad_kernel<<<(G_*KP1  + 255)/256, 256, 0, stream>>>(Wih3, Wi3, G_, KP1, KP1);
  cvt_wout_kernel<<<(128*KPO + 255)/256, 256, 0, stream>>>(Wout, Wob);

  const size_t smem1  = (size_t)64*KP1*2 + 16384 + 4096 + 4*32*4 + 4*16*4 + 4*64*4;
  const size_t smem23 = (size_t)64*KP23*2 + 16384 + 4096;

  // Layer 1 (persistent, attention inside)
  lstm1_kernel<<<NWG, 256, smem1, stream>>>(
      x, text, mask, Wc1, Wih1, b1, Wwin, bwin, kap, h1b, wbuf, A1a, A1b, bar);

  // Layer 2: input GEMM then persistent recurrence
  build_xcat_kernel<<<((long)BT*KP1 + 255)/256, 256, 0, stream>>>(x, h1b, wbuf, Xcat);
  gemm_nt_kernel<<<dim3(BT/256, G_/64), 256, 0, stream>>>(Xcat, Wi2, pre, G_, KP1, 15);
  lstm_rnn_kernel<<<NWG, 256, smem23, stream>>>(pre, Wh2, b2, h2b, A2a, A2b, bar);

  // Layer 3
  build_xcat_kernel<<<((long)BT*KP1 + 255)/256, 256, 0, stream>>>(x, h2b, wbuf, Xcat);
  gemm_nt_kernel<<<dim3(BT/256, G_/64), 256, 0, stream>>>(Xcat, Wi3, pre, G_, KP1, 15);
  lstm_rnn_kernel<<<NWG, 256, smem23, stream>>>(pre, Wh3, b3, h3b, A3a, A3b, bar);

  // Output projection (N padded to 128)
  gemm_out_kernel<<<dim3(BT/256, 2), 256, 0, stream>>>(h1b, h2b, h3b, Wob, bout, out);

  (void)in_sizes; (void)n_in; (void)out_size; (void)ws_size;
}
